// GraphDistillationLoss_37237366456558
// MI455X (gfx1250) — compile-verified, matrix operand order checked by reading, not run on hardware
//
#include <hip/hip_runtime.h>
#include <hip/hip_bf16.h>

#define NN 4096
#define DIM 512
#define EPS_GW 0.05f
#define SINK_ITERS 100
#define TINYF 1e-12f
#define INV_N (1.0f/4096.0f)

typedef float  v2f   __attribute__((ext_vector_type(2)));
typedef float  v8f   __attribute__((ext_vector_type(8)));
typedef __bf16 v16bf __attribute__((ext_vector_type(16)));
typedef __bf16 v8bf  __attribute__((ext_vector_type(8)));
typedef __bf16 v4bf  __attribute__((ext_vector_type(4)));

// ---------------- helpers ----------------

__device__ inline float block_reduce256(float val, float* red) {
  int t = threadIdx.x;
  red[t] = val; __syncthreads();
  for (int o = 128; o > 0; o >>= 1) {
    if (t < o) red[t] += red[t + o];
    __syncthreads();
  }
  float r = red[0];
  __syncthreads();
  return r;
}

// Async copy of one 128x32 bf16 tile (row-major, ld = K) into LDS.
// 512 chunks of 16B; 256 threads -> 2 chunks each. ASYNCcnt-tracked.
__device__ inline void stage_tile_async(const __bf16* __restrict__ G, int ld,
                                        int rowBase, int kb,
                                        __bf16* lbuf, int tid) {
#pragma unroll
  for (int c = tid; c < 512; c += 256) {
    int row = c >> 2;
    int kk = (c & 3) * 8;
    uint64_t gaddr = (uint64_t)(uintptr_t)(G + (size_t)(rowBase + row) * ld + kb + kk);
    uint32_t lOff = (uint32_t)(uintptr_t)(lbuf + row * 32 + kk);
    asm volatile("global_load_async_to_lds_b128 %0, %1, off"
                 :: "v"(lOff), "v"(gaddr) : "memory");
  }
}

__device__ inline void wait_async_all() {
  asm volatile("s_wait_asynccnt 0x0" ::: "memory");
}

// ---------------- normalize ----------------

__global__ __launch_bounds__(256) void k_normalize(const float* __restrict__ in,
                                                   float* __restrict__ out) {
  __shared__ float red[256];
  int row = blockIdx.x;
  const float* r = in + (size_t)row * DIM;
  float s = 0.f;
  for (int c = threadIdx.x; c < DIM; c += 256) { float x = r[c]; s += x * x; }
  float tot = block_reduce256(s, red);
  float n = fmaxf(sqrtf(tot), TINYF);
  float inv = 1.0f / n;
  float* w = out + (size_t)row * DIM;
  for (int c = threadIdx.x; c < DIM; c += 256) w[c] = r[c] * inv;
}

// ---------------- f32 WMMA NT GEMM: C = A * B^T ----------------
// A [M x K] row-major, B [N x K] row-major. Block tile 128x128, wave tile 32x64.
// EPI 0: store raw f32.  EPI 1: store exp((acc-1)/eps)  (Sinkhorn kernel K).

template <int EPI>
__global__ __launch_bounds__(256) void k_gemm_f32_nt(const float* __restrict__ A,
                                                     const float* __restrict__ B,
                                                     float* __restrict__ C,
                                                     int K, int ldc) {
  int lane = threadIdx.x & 31;
  int wave = threadIdx.x >> 5;
  int wm = wave & 3, wn = wave >> 2;
  int r = lane & 15, g = lane >> 4;
  int m0 = blockIdx.y * 128 + wm * 32;
  int n0 = blockIdx.x * 128 + wn * 64;

  v8f acc[2][4] = {};
  const float* Ap = A + (size_t)(m0 + r) * K + 2 * g;
  const float* Bp = B + (size_t)(n0 + r) * K + 2 * g;

  for (int k = 0; k < K; k += 4) {
    v2f a[2], b[4];
#pragma unroll
    for (int mi = 0; mi < 2; mi++) a[mi] = *(const v2f*)(Ap + (size_t)mi * 16 * K + k);
#pragma unroll
    for (int ni = 0; ni < 4; ni++) b[ni] = *(const v2f*)(Bp + (size_t)ni * 16 * K + k);
#pragma unroll
    for (int mi = 0; mi < 2; mi++)
#pragma unroll
      for (int ni = 0; ni < 4; ni++)
        acc[mi][ni] = __builtin_amdgcn_wmma_f32_16x16x4_f32(
            false, a[mi], false, b[ni], (short)0, acc[mi][ni], false, false);
  }

#pragma unroll
  for (int mi = 0; mi < 2; mi++)
#pragma unroll
    for (int ni = 0; ni < 4; ni++)
#pragma unroll
      for (int e = 0; e < 8; e++) {
        int gi = m0 + mi * 16 + e + 8 * g;
        int gj = n0 + ni * 16 + r;
        float val = acc[mi][ni][e];
        if (EPI == 1) val = expf((val - 1.0f) * (1.0f / EPS_GW));
        C[(size_t)gi * ldc + gj] = val;
      }
}

// ---------------- bf16 WMMA NT GEMM with async-LDS staging ----------------
// C = A * B^T. A [M x K], B [N x K] bf16 row-major, fp32 accumulate.
// Double-buffered 128x128x32 tiles staged via global_load_async_to_lds_b128.
// EPI 0: store f32 C (ldc = NN).
// EPI 1: fused L_edge: sum (Ss_ij - P_ij)^2 with diagonal forced to (Ss_ii - 1)^2.
// EPI 2: fused L_gw:   sum (max(1-Ss_ij,0) - P_ij)^2.

template <int EPI>
__global__ __launch_bounds__(256) void k_gemm_bf16_nt(const __bf16* __restrict__ A,
                                                      const __bf16* __restrict__ B,
                                                      float* __restrict__ C,
                                                      const float* __restrict__ Ref,
                                                      float* __restrict__ lossPartial,
                                                      int K) {
  __shared__ float red[256];
  __shared__ __align__(16) __bf16 shA[2][128 * 32];
  __shared__ __align__(16) __bf16 shB[2][128 * 32];

  int tid = threadIdx.x;
  int lane = tid & 31;
  int wave = tid >> 5;
  int wm = wave & 3, wn = wave >> 2;
  int r = lane & 15, g = lane >> 4;
  int mBlk = blockIdx.y * 128;
  int nBlk = blockIdx.x * 128;

  v8f acc[2][4] = {};

  // prologue: stage k=0 into buffer 0
  stage_tile_async(A, K, mBlk, 0, &shA[0][0], tid);
  stage_tile_async(B, K, nBlk, 0, &shB[0][0], tid);

  int buf = 0;
  for (int kb = 0; kb < K; kb += 32) {
    wait_async_all();       // this wave's async writes to shA/shB[buf] landed
    __syncthreads();        // all waves' writes visible; prior reads of buf^1 done
    if (kb + 32 < K) {
      stage_tile_async(A, K, mBlk, kb + 32, &shA[buf ^ 1][0], tid);
      stage_tile_async(B, K, nBlk, kb + 32, &shB[buf ^ 1][0], tid);
    }
    v16bf a[2], b[4];
#pragma unroll
    for (int mi = 0; mi < 2; mi++) {
      union { v16bf v; v8bf h[2]; } u;
      const __bf16* p = &shA[buf][(wm * 32 + mi * 16 + r) * 32 + 8 * g];
      u.h[0] = *(const v8bf*)(p);
      u.h[1] = *(const v8bf*)(p + 16);
      a[mi] = u.v;
    }
#pragma unroll
    for (int ni = 0; ni < 4; ni++) {
      union { v16bf v; v8bf h[2]; } u;
      const __bf16* p = &shB[buf][(wn * 64 + ni * 16 + r) * 32 + 8 * g];
      u.h[0] = *(const v8bf*)(p);
      u.h[1] = *(const v8bf*)(p + 16);
      b[ni] = u.v;
    }
#pragma unroll
    for (int mi = 0; mi < 2; mi++)
#pragma unroll
      for (int ni = 0; ni < 4; ni++)
        acc[mi][ni] = __builtin_amdgcn_wmma_f32_16x16x32_bf16(
            false, a[mi], false, b[ni], (short)0, acc[mi][ni], false, false);
    buf ^= 1;
  }

  if (EPI == 0) {
#pragma unroll
    for (int mi = 0; mi < 2; mi++)
#pragma unroll
      for (int ni = 0; ni < 4; ni++)
#pragma unroll
        for (int e = 0; e < 8; e++) {
          int gi = mBlk + wm * 32 + mi * 16 + e + 8 * g;
          int gj = nBlk + wn * 64 + ni * 16 + r;
          C[(size_t)gi * NN + gj] = acc[mi][ni][e];
        }
  } else {
    float ls = 0.f;
#pragma unroll
    for (int mi = 0; mi < 2; mi++)
#pragma unroll
      for (int ni = 0; ni < 4; ni++)
#pragma unroll
        for (int e = 0; e < 8; e++) {
          int gi = mBlk + wm * 32 + mi * 16 + e + 8 * g;
          int gj = nBlk + wn * 64 + ni * 16 + r;
          float pv = acc[mi][ni][e];
          float ref = Ref[(size_t)gi * NN + gj];
          float d;
          if (EPI == 1) {
            d = (gi == gj) ? (ref - 1.0f) : (ref - pv);
          } else {
            float ds = fmaxf(1.0f - ref, 0.0f);
            d = ds - pv;
          }
          ls += d * d;
        }
    float tot = block_reduce256(ls, red);
    if (threadIdx.x == 0)
      lossPartial[blockIdx.y * gridDim.x + blockIdx.x] = tot;
  }
}

// ---------------- Sinkhorn ----------------

__global__ __launch_bounds__(256) void k_fill(float* p, float val, int n) {
  int i = blockIdx.x * 256 + threadIdx.x;
  if (i < n) p[i] = val;
}

// u[i] = (1/N) / (sum_j K[i][j] v[j] + TINY)
__global__ __launch_bounds__(256) void k_rowmv(const float* __restrict__ Km,
                                               const float* __restrict__ v,
                                               float* __restrict__ u) {
  __shared__ float red[256];
  int row = blockIdx.x;
  const float* kr = Km + (size_t)row * NN;
  float s = 0.f;
  for (int j = threadIdx.x; j < NN; j += 256) s += kr[j] * v[j];
  float tot = block_reduce256(s, red);
  if (threadIdx.x == 0) u[row] = INV_N / (tot + TINYF);
}

// partial[by][j] = sum_{i in 256-chunk by} K[i][j] * u[i]   (deterministic 2-stage)
__global__ __launch_bounds__(256) void k_colpartial(const float* __restrict__ Km,
                                                    const float* __restrict__ u,
                                                    float* __restrict__ part) {
  int j = blockIdx.x * 256 + threadIdx.x;
  int i0 = blockIdx.y * 256;
  float s = 0.f;
  for (int i = i0; i < i0 + 256; i++) s += Km[(size_t)i * NN + j] * u[i];
  part[(size_t)blockIdx.y * NN + j] = s;
}

__global__ __launch_bounds__(256) void k_colfinal(const float* __restrict__ part,
                                                  float* __restrict__ v) {
  int j = blockIdx.x * 256 + threadIdx.x;
  float s = 0.f;
  for (int p = 0; p < 16; p++) s += part[(size_t)p * NN + j];
  v[j] = INV_N / (s + TINYF);
}

// pi = u[i] * K * v[j], in place (float4 vectorized; 4 elems share a row)
__global__ __launch_bounds__(256) void k_pi(float* __restrict__ Km,
                                            const float* __restrict__ u,
                                            const float* __restrict__ v) {
  size_t base = ((size_t)blockIdx.x * 256 + threadIdx.x) * 4;
  int i = (int)(base >> 12);
  int j = (int)(base & 4095);
  float4 kv = *(float4*)(Km + base);
  float ui = u[i];
  kv.x *= ui * v[j + 0];
  kv.y *= ui * v[j + 1];
  kv.z *= ui * v[j + 2];
  kv.w *= ui * v[j + 3];
  *(float4*)(Km + base) = kv;
}

__global__ __launch_bounds__(256) void k_rowsum(const float* __restrict__ Pm,
                                                float* __restrict__ rowsum) {
  __shared__ float red[256];
  int row = blockIdx.x;
  const float* pr = Pm + (size_t)row * NN;
  float s = 0.f;
  for (int j = threadIdx.x; j < NN; j += 256) s += pr[j];
  float tot = block_reduce256(s, red);
  if (threadIdx.x == 0) rowsum[row] = fmaxf(tot, TINYF);
}

// ---------------- conversions / transpose ----------------

__global__ __launch_bounds__(256) void k_cvt_bf16(const float* __restrict__ in,
                                                  __bf16* __restrict__ out) {
  size_t base = ((size_t)blockIdx.x * 256 + threadIdx.x) * 4;
  float4 x = *(const float4*)(in + base);
  v4bf y;
  y[0] = (__bf16)x.x; y[1] = (__bf16)x.y; y[2] = (__bf16)x.z; y[3] = (__bf16)x.w;
  *(v4bf*)(out + base) = y;
}

// D_t = max(1 - S_t, 0) -> bf16
__global__ __launch_bounds__(256) void k_cvt_dt(const float* __restrict__ in,
                                                __bf16* __restrict__ out) {
  size_t base = ((size_t)blockIdx.x * 256 + threadIdx.x) * 4;
  float4 x = *(const float4*)(in + base);
  v4bf y;
  y[0] = (__bf16)fmaxf(1.0f - x.x, 0.0f);
  y[1] = (__bf16)fmaxf(1.0f - x.y, 0.0f);
  y[2] = (__bf16)fmaxf(1.0f - x.z, 0.0f);
  y[3] = (__bf16)fmaxf(1.0f - x.w, 0.0f);
  *(v4bf*)(out + base) = y;
}

// transpose [NN x DIM] -> [DIM x NN] (for the pi @ t_norm GEMM as NT)
__global__ __launch_bounds__(256) void k_transpose(const float* __restrict__ in,
                                                   float* __restrict__ out) {
  __shared__ float tile[32][33];
  int tx = threadIdx.x & 31, ty = threadIdx.x >> 5;
  int c0 = blockIdx.x * 32;   // over DIM
  int r0 = blockIdx.y * 32;   // over NN
  for (int yy = ty; yy < 32; yy += 8)
    tile[yy][tx] = in[(size_t)(r0 + yy) * DIM + c0 + tx];
  __syncthreads();
  for (int yy = ty; yy < 32; yy += 8)
    out[(size_t)(c0 + yy) * NN + r0 + tx] = tile[tx][yy];
}

// ---------------- L_node ----------------

__global__ __launch_bounds__(256) void k_node_loss(const float* __restrict__ sn,
                                                   const float* __restrict__ ts_raw,
                                                   const float* __restrict__ rowsum,
                                                   float* __restrict__ nodep) {
  __shared__ float red[256];
  int row = blockIdx.x;
  float inv = 1.0f / rowsum[row];
  float s = 0.f;
  for (int c = threadIdx.x; c < DIM; c += 256) {
    float d = sn[(size_t)row * DIM + c] - ts_raw[(size_t)row * DIM + c] * inv;
    s += d * d;
  }
  float tot = block_reduce256(s, red);
  if (threadIdx.x == 0) nodep[row] = tot;
}

// ---------------- final combine ----------------

__global__ __launch_bounds__(256) void k_final(const float* __restrict__ nodep,
                                               const float* __restrict__ edgep,
                                               const float* __restrict__ gwp,
                                               float* __restrict__ out) {
  __shared__ float red[256];
  int t = threadIdx.x;
  float s = 0.f;
  for (int i = t; i < NN; i += 256) s += nodep[i];
  float Ln = block_reduce256(s, red) / ((float)NN * (float)DIM);
  s = 0.f;
  for (int i = t; i < 1024; i += 256) s += edgep[i];
  float Le = block_reduce256(s, red) / ((float)NN * (float)NN);
  s = 0.f;
  for (int i = t; i < 1024; i += 256) s += gwp[i];
  float Lg = block_reduce256(s, red) / ((float)NN * (float)NN);
  if (t == 0) out[0] = Ln + Le + Lg;
}

// ---------------- host ----------------

extern "C" void kernel_launch(void* const* d_in, const int* in_sizes, int n_in,
                              void* d_out, int out_size, void* d_ws, size_t ws_size,
                              hipStream_t stream) {
  (void)in_sizes; (void)n_in; (void)out_size; (void)ws_size;
  const float* s_nodes = (const float*)d_in[0];
  const float* t_nodes = (const float*)d_in[1];
  float* out = (float*)d_out;

  constexpr size_t SZ_ND = (size_t)NN * DIM * 4;      // 8 MB
  constexpr size_t SZ_NN = (size_t)NN * NN * 4;       // 64 MB
  constexpr size_t SZ_NN_BF = (size_t)NN * NN * 2;    // 32 MB

  char* w = (char*)d_ws;
  float* sn    = (float*)w;              w += SZ_ND;
  float* tn    = (float*)w;              w += SZ_ND;
  float* tnT   = (float*)w;              w += SZ_ND;
  float* tsoft = (float*)w;              w += SZ_ND;
  float* Km    = (float*)w;              w += SZ_NN;  // K, then pi in-place
  float* Ss    = (float*)w;              w += SZ_NN;
  float* St    = (float*)w;              w += SZ_NN;
  float* M1    = (float*)w;              w += SZ_NN;
  __bf16* pib  = (__bf16*)w;             w += SZ_NN_BF;
  __bf16* bb   = (__bf16*)w;             w += SZ_NN_BF;  // S_t / D_t bf16
  __bf16* ab   = (__bf16*)w;             w += SZ_NN_BF;  // M1 bf16
  float* uvec  = (float*)w;              w += NN * 4;
  float* vvec  = (float*)w;              w += NN * 4;
  float* part  = (float*)w;              w += (size_t)16 * NN * 4;
  float* rowsum= (float*)w;              w += NN * 4;
  float* nodep = (float*)w;              w += NN * 4;
  float* edgep = (float*)w;              w += 1024 * 4;
  float* gwp   = (float*)w;              w += 1024 * 4;

  dim3 blk(256);
  dim3 gNN(32, 32);           // 128x128 tiles over 4096x4096
  dim3 gTS(4, 32);            // N=512, M=4096
  const int cvtBlocks = (int)((size_t)NN * NN / 4 / 256);  // 16384

  // 1) normalize
  k_normalize<<<NN, blk, 0, stream>>>(s_nodes, sn);
  k_normalize<<<NN, blk, 0, stream>>>(t_nodes, tn);

  // 2) gram GEMMs (f32 WMMA): S_s, S_t, K = exp((S_st - 1)/eps)
  k_gemm_f32_nt<0><<<gNN, blk, 0, stream>>>(sn, sn, Ss, DIM, NN);
  k_gemm_f32_nt<0><<<gNN, blk, 0, stream>>>(tn, tn, St, DIM, NN);
  k_gemm_f32_nt<1><<<gNN, blk, 0, stream>>>(sn, tn, Km, DIM, NN);

  // 3) Sinkhorn (K is L2-resident: 64 MB < 192 MB)
  k_fill<<<16, blk, 0, stream>>>(vvec, 1.0f, NN);
  for (int it = 0; it < SINK_ITERS; it++) {
    k_rowmv<<<NN, blk, 0, stream>>>(Km, vvec, uvec);
    k_colpartial<<<dim3(16, 16), blk, 0, stream>>>(Km, uvec, part);
    k_colfinal<<<16, blk, 0, stream>>>(part, vvec);
  }

  // 4) pi = u * K * v (in place), row sums, bf16 copy of pi
  k_pi<<<cvtBlocks, blk, 0, stream>>>(Km, uvec, vvec);
  k_rowsum<<<NN, blk, 0, stream>>>(Km, rowsum);
  k_cvt_bf16<<<cvtBlocks, blk, 0, stream>>>(Km, pib);

  // 5) L_node: tsoft = pi @ t_norm via NT GEMM against t_norm^T (f32 WMMA)
  k_transpose<<<dim3(16, 128), blk, 0, stream>>>(tn, tnT);
  k_gemm_f32_nt<0><<<gTS, blk, 0, stream>>>(Km, tnT, tsoft, NN, DIM);
  k_node_loss<<<NN, blk, 0, stream>>>(sn, tsoft, rowsum, nodep);

  // 6) L_edge: P = (pi @ S_t) @ pi^T, fused squared-diff vs S_s (diag -> 1)
  k_cvt_bf16<<<cvtBlocks, blk, 0, stream>>>(St, bb);
  k_gemm_bf16_nt<0><<<gNN, blk, 0, stream>>>(pib, bb, M1, nullptr, nullptr, NN);
  k_cvt_bf16<<<cvtBlocks, blk, 0, stream>>>(M1, ab);
  k_gemm_bf16_nt<1><<<gNN, blk, 0, stream>>>(ab, pib, nullptr, Ss, edgep, NN);

  // 7) L_gw: same with D_t = max(1 - S_t, 0), compared vs D_s = max(1 - S_s, 0)
  k_cvt_dt<<<cvtBlocks, blk, 0, stream>>>(St, bb);
  k_gemm_bf16_nt<0><<<gNN, blk, 0, stream>>>(pib, bb, M1, nullptr, nullptr, NN);
  k_cvt_bf16<<<cvtBlocks, blk, 0, stream>>>(M1, ab);
  k_gemm_bf16_nt<2><<<gNN, blk, 0, stream>>>(ab, pib, nullptr, Ss, gwp, NN);

  // 8) combine
  k_final<<<1, blk, 0, stream>>>(nodep, edgep, gwp, out);
}
